// LSTMClassifier_36713380446870
// MI455X (gfx1250) — matrix operand order repaired and checked by reading, not compiled, not run on hardware
//
#include <hip/hip_runtime.h>

// ---------------- problem constants ----------------
static constexpr int Bn = 64;     // batch
static constexpr int Tn = 512;    // seq len
static constexpr int En = 512;    // embed dim
static constexpr int Hn = 1024;   // hidden dim
static constexpr int G4 = 4 * Hn; // 4096 gate rows
static constexpr int On = 8;      // output dim
static constexpr int PAD = 1;

typedef __attribute__((ext_vector_type(16))) __bf16 v16bf;
typedef __attribute__((ext_vector_type(8)))  float  v8f;

// ---------------- workspace layout (bytes) ----------------
static constexpr size_t OFF_WIH   = 0;                                   // [4096][512]  bf16  4 MB
static constexpr size_t OFF_WHH   = OFF_WIH  + (size_t)G4 * En * 2;      // [4096][1024] bf16  8 MB
static constexpr size_t OFF_EMB   = OFF_WHH  + (size_t)G4 * Hn * 2;      // [T][B][E]    bf16 32 MB
static constexpr size_t OFF_H0    = OFF_EMB  + (size_t)Tn * Bn * En * 2; // [B][H] bf16 128 KB
static constexpr size_t OFF_H1    = OFF_H0   + (size_t)Bn * Hn * 2;
static constexpr size_t OFF_C     = OFF_H1   + (size_t)Bn * Hn * 2;      // [B][H] f32
static constexpr size_t OFF_FINAL = OFF_C    + (size_t)Bn * Hn * 4;      // [B][H] f32
static constexpr size_t OFF_BIAS  = OFF_FINAL+ (size_t)Bn * Hn * 4;      // [4096] f32
static constexpr size_t OFF_FINT  = OFF_BIAS + (size_t)G4 * 4;           // [B] int
static constexpr size_t OFF_BAR   = OFF_FINT + 256;                      // bar_cnt[4], bar_gen[4]

// ---------------- helpers ----------------
static __device__ __forceinline__ float sigf(float x) {
    return 1.0f / (1.0f + __expf(-x));
}
static __device__ __forceinline__ float tanh_fast(float x) {
    // tanh(x) = 1 - 2/(exp(2x)+1); saturates correctly at +-inf
    return 1.0f - 2.0f / (__expf(2.0f * x) + 1.0f);
}

// A fragment: 16x32 bf16, lane L holds row M = L%16; VGPR0-3 = K + 8*(L/16) .. +7,
// VGPR4-7 = K + 16 + 8*(L/16) .. +7  -> two b128 loads
static __device__ __forceinline__ v16bf load_a_frag(const __bf16* __restrict__ row,
                                                    int K0, int khalf) {
    const __bf16* p = row + K0 + khalf * 8;
    union { uint4 u[2]; v16bf v; } r;
    r.u[0] = *(const uint4*)(p);
    r.u[1] = *(const uint4*)(p + 16);
    return r.v;
}
// B fragment: 32x16 bf16, lane L holds col N = L%16; VGPR0-3 = K + 8*(L/16) .. +7, VGPR4-7 = K + 16 + 8*(L/16) .. +7
// weights row-major [N][K] -> K contiguous -> two b128 loads
static __device__ __forceinline__ v16bf load_b_frag(const __bf16* __restrict__ wrow,
                                                    int K0, int khalf) {
    const __bf16* p = wrow + K0 + khalf * 8;
    union { uint4 u[2]; v16bf v; } r;
    r.u[0] = *(const uint4*)(p);
    r.u[1] = *(const uint4*)(p + 16);
    return r.v;
}
static __device__ __forceinline__ v8f wmma_bf16(v16bf a, v16bf b, v8f c) {
    return __builtin_amdgcn_wmma_f32_16x16x32_bf16(false, a, false, b,
                                                   (short)0, c, false, false);
}

// ---------------- prep kernels ----------------
__global__ void cvt_bf16_kernel(const float* __restrict__ src, __bf16* __restrict__ dst, int n) {
    int i = blockIdx.x * blockDim.x + threadIdx.x;
    if (i < n) dst[i] = (__bf16)src[i];
}

__global__ void bias_kernel(const float* __restrict__ bih, const float* __restrict__ bhh,
                            float* __restrict__ bias) {
    int i = blockIdx.x * blockDim.x + threadIdx.x;
    if (i < G4) bias[i] = bih[i] + bhh[i];
}

__global__ void init_kernel(__bf16* __restrict__ h0, float* __restrict__ c,
                            float* __restrict__ fin, int* __restrict__ bar) {
    int i = blockIdx.x * blockDim.x + threadIdx.x;
    if (i < Bn * Hn) { h0[i] = (__bf16)0.0f; c[i] = 0.0f; fin[i] = 0.0f; }
    if (i < 8) bar[i] = 0;   // bar_cnt[0..3], bar_gen[0..3]
}

__global__ void len_kernel(const int* __restrict__ x, int* __restrict__ fint) {
    int b = threadIdx.x;
    if (b < Bn) {
        int cnt = 0;
        for (int t = 0; t < Tn; ++t) cnt += (x[b * Tn + t] != PAD);
        fint[b] = (cnt + Tn - 1) % Tn;   // len-1, with Python -1 wrap when len==0
    }
}

// gather embeddings -> [T][B][E] bf16
__global__ void emb_gather_kernel(const int* __restrict__ x, const float* __restrict__ Wemb,
                                  __bf16* __restrict__ embT) {
    size_t i = (size_t)blockIdx.x * blockDim.x + threadIdx.x;   // over T*B*E
    int e = (int)(i % En);
    int b = (int)((i / En) % Bn);
    int t = (int)(i / ((size_t)En * Bn));
    int idx = x[b * Tn + t];
    embT[i] = (__bf16)Wemb[(size_t)idx * En + e];
}

// ---------------- persistent fused LSTM ----------------
// 32 blocks x 256 threads = 256 waves. Group g = blockIdx/8 owns batch tile mt = g
// (16 batch rows); its 8 blocks x 8 waves cover jt = 0..63 hidden-unit tiles.
// Cross-step dependency exists ONLY within a group (h rows of one batch tile), so
// each group runs all 512 timesteps with a device-side 8-block barrier — no global
// sync, no per-step kernel launches.
__global__ __launch_bounds__(256)
void lstm_persistent_kernel(const __bf16* __restrict__ embT,  // [T][B][E]
                            const __bf16* __restrict__ Wih,   // [4096][512]
                            const __bf16* __restrict__ Whh,   // [4096][1024]
                            const float*  __restrict__ bias,  // [4096]
                            const int*    __restrict__ x,     // [B][T]
                            const int*    __restrict__ fint,  // [B]
                            __bf16*       __restrict__ h0,    // [B][H] bf16 (even-t in)
                            __bf16*       __restrict__ h1,    // [B][H] bf16 (odd-t in)
                            float*        __restrict__ c,     // [B][H]
                            float*        __restrict__ finalh,// [B][H]
                            int*          __restrict__ bar_cnt,
                            int*          __restrict__ bar_gen) {
    const int tid   = threadIdx.x;
    const int lane  = tid & 31;
    const int mt    = blockIdx.x >> 3;                     // group = batch tile 0..3
    const int jt    = (blockIdx.x & 7) * 8 + (tid >> 5);   // hidden-unit tile 0..63
    const int khalf = lane >> 4;
    const int ncol  = lane & 15;
    const int m     = mt * 16 + ncol;    // A row for this lane
    const int jg    = jt * 16 + ncol;    // gate column (within H)

    // loop-invariant per-lane state
    const __bf16* wi_i = Wih + (size_t)(0 * Hn + jg) * En;
    const __bf16* wi_f = Wih + (size_t)(1 * Hn + jg) * En;
    const __bf16* wi_g = Wih + (size_t)(2 * Hn + jg) * En;
    const __bf16* wi_o = Wih + (size_t)(3 * Hn + jg) * En;
    const __bf16* wh_i = Whh + (size_t)(0 * Hn + jg) * Hn;
    const __bf16* wh_f = Whh + (size_t)(1 * Hn + jg) * Hn;
    const __bf16* wh_g = Whh + (size_t)(2 * Hn + jg) * Hn;
    const __bf16* wh_o = Whh + (size_t)(3 * Hn + jg) * Hn;
    const float bi = bias[0 * Hn + jg];
    const float bf = bias[1 * Hn + jg];
    const float bg = bias[2 * Hn + jg];
    const float bo = bias[3 * Hn + jg];
    int fintv[8];
#pragma unroll
    for (int r = 0; r < 8; ++r) fintv[r] = fint[mt * 16 + khalf * 8 + r];

    const __bf16* arow_e = embT + (size_t)m * En;   // advances by Bn*En per step
    const __bf16* hrow0  = h0 + (size_t)m * Hn;
    const __bf16* hrow1  = h1 + (size_t)m * Hn;

    for (int t = 0; t < Tn; ++t) {
        const __bf16* arow_h = (t & 1) ? hrow1 : hrow0;
        __bf16*       hout   = (t & 1) ? h0 : h1;

        v8f acc_i = {}, acc_f = {}, acc_g = {}, acc_o = {};

        // input projection: K = 0..511 over embeddings
        for (int K0 = 0; K0 < En; K0 += 32) {
            v16bf a = load_a_frag(arow_e, K0, khalf);
            acc_i = wmma_bf16(a, load_b_frag(wi_i, K0, khalf), acc_i);
            acc_f = wmma_bf16(a, load_b_frag(wi_f, K0, khalf), acc_f);
            acc_g = wmma_bf16(a, load_b_frag(wi_g, K0, khalf), acc_g);
            acc_o = wmma_bf16(a, load_b_frag(wi_o, K0, khalf), acc_o);
        }
        // recurrence: K = 0..1023 over h_{t-1}
        for (int K0 = 0; K0 < Hn; K0 += 32) {
            v16bf a = load_a_frag(arow_h, K0, khalf);
            acc_i = wmma_bf16(a, load_b_frag(wh_i, K0, khalf), acc_i);
            acc_f = wmma_bf16(a, load_b_frag(wh_f, K0, khalf), acc_f);
            acc_g = wmma_bf16(a, load_b_frag(wh_g, K0, khalf), acc_g);
            acc_o = wmma_bf16(a, load_b_frag(wh_o, K0, khalf), acc_o);
        }

        // gates + state update (each (mb,jg) element owned by exactly one lane)
#pragma unroll
        for (int r = 0; r < 8; ++r) {
            const int mb = mt * 16 + khalf * 8 + r;   // C/D row = r + 8*(lane/16)
            const size_t idx = (size_t)mb * Hn + jg;
            const float zi = acc_i[r] + bi;
            const float zf = acc_f[r] + bf;
            const float zg = acc_g[r] + bg;
            const float zo = acc_o[r] + bo;
            const float cn = sigf(zf) * c[idx] + sigf(zi) * tanh_fast(zg);
            const float hn = sigf(zo) * tanh_fast(cn);
            c[idx]    = cn;
            hout[idx] = (__bf16)hn;
            if (t == fintv[r]) {
                const bool pad = (x[mb * Tn + t] == PAD);
                finalh[idx] = pad ? 0.0f : hn;
            }
        }

        // ---- group barrier: 8 blocks of this batch-tile group ----
        __threadfence();            // release h/c stores to device scope
        __syncthreads();
        if (tid == 0) {
            int prev = __hip_atomic_fetch_add(&bar_cnt[mt], 1,
                                              __ATOMIC_ACQ_REL, __HIP_MEMORY_SCOPE_AGENT);
            if ((prev & 7) == 7) {  // last arrival of this generation
                __hip_atomic_store(&bar_gen[mt], t + 1,
                                   __ATOMIC_RELEASE, __HIP_MEMORY_SCOPE_AGENT);
            } else {
                while (__hip_atomic_load(&bar_gen[mt],
                                         __ATOMIC_ACQUIRE, __HIP_MEMORY_SCOPE_AGENT) <= t)
                    __builtin_amdgcn_s_sleep(2);
            }
        }
        __syncthreads();
        __threadfence();            // acquire: invalidate stale h in near caches

        arow_e += (size_t)Bn * En;
    }
}

// ---------------- classifier head: logits = final @ W_out^T + b_out ----------------
__global__ void out_kernel(const float* __restrict__ finalh, const float* __restrict__ Wout,
                           const float* __restrict__ bout, float* __restrict__ out) {
    const int b = blockIdx.x;          // 64 blocks
    const int o = threadIdx.x >> 5;    // 8 waves -> 8 outputs
    const int lane = threadIdx.x & 31;
    const float* f = finalh + (size_t)b * Hn;
    const float* w = Wout + (size_t)o * Hn;
    float s = 0.0f;
    for (int k = lane; k < Hn; k += 32) s += f[k] * w[k];
    for (int off = 16; off > 0; off >>= 1) s += __shfl_down(s, off, 32);
    if (lane == 0) out[b * On + o] = s + bout[o];
}

// ---------------- launcher ----------------
extern "C" void kernel_launch(void* const* d_in, const int* in_sizes, int n_in,
                              void* d_out, int out_size, void* d_ws, size_t ws_size,
                              hipStream_t stream) {
    const int*   x    = (const int*)d_in[0];
    const float* Wemb = (const float*)d_in[1];
    const float* Wih  = (const float*)d_in[2];
    const float* Whh  = (const float*)d_in[3];
    const float* bih  = (const float*)d_in[4];
    const float* bhh  = (const float*)d_in[5];
    const float* Wout = (const float*)d_in[6];
    const float* bout = (const float*)d_in[7];
    float* out = (float*)d_out;

    char* ws = (char*)d_ws;
    __bf16* wih_b  = (__bf16*)(ws + OFF_WIH);
    __bf16* whh_b  = (__bf16*)(ws + OFF_WHH);
    __bf16* embT   = (__bf16*)(ws + OFF_EMB);
    __bf16* h0     = (__bf16*)(ws + OFF_H0);
    __bf16* h1     = (__bf16*)(ws + OFF_H1);
    float*  c      = (float*)(ws + OFF_C);
    float*  finalh = (float*)(ws + OFF_FINAL);
    float*  bias   = (float*)(ws + OFF_BIAS);
    int*    fint   = (int*)(ws + OFF_FINT);
    int*    bar    = (int*)(ws + OFF_BAR);      // [0..3]=cnt, [4..7]=gen

    // prep: bf16 casts, state + barrier init, final-step indices, embedding gather
    cvt_bf16_kernel<<<(G4 * En + 255) / 256, 256, 0, stream>>>(Wih, wih_b, G4 * En);
    cvt_bf16_kernel<<<(G4 * Hn + 255) / 256, 256, 0, stream>>>(Whh, whh_b, G4 * Hn);
    bias_kernel<<<(G4 + 255) / 256, 256, 0, stream>>>(bih, bhh, bias);
    init_kernel<<<(Bn * Hn + 255) / 256, 256, 0, stream>>>(h0, c, finalh, bar);
    len_kernel<<<1, 64, 0, stream>>>(x, fint);
    emb_gather_kernel<<<(int)(((size_t)Tn * Bn * En) / 256), 256, 0, stream>>>(x, Wemb, embT);

    // one persistent kernel runs all 512 timesteps (device-side group barriers)
    lstm_persistent_kernel<<<32, 256, 0, stream>>>(embT, wih_b, whh_b, bias, x, fint,
                                                   h0, h1, c, finalh, bar, bar + 4);

    out_kernel<<<Bn, 256, 0, stream>>>(finalh, Wout, bout, out);
}
